// DBFLinear_GEMLITE_30124900614507
// MI455X (gfx1250) — matrix operand order repaired and checked by reading, not compiled, not run on hardware
//
#include <hip/hip_runtime.h>
#include <cstdint>
#include <cstddef>

typedef __attribute__((ext_vector_type(16))) __bf16 v16bf;
typedef __attribute__((ext_vector_type(8)))  float  v8f;
typedef __attribute__((ext_vector_type(4)))  int    v4i;
typedef unsigned short u16;

#define AS1 __attribute__((address_space(1)))
#define AS3 __attribute__((address_space(3)))

#define M_TOT   8192
#define K_DIM   4096
#define N_DIM   4096

#define BM 128
#define BN 128
#define BK 64
#define LDA 72            // padded row (bf16 elems): 144B, 16B aligned, bank-rotated
#define NTHREADS 256

union FragAB { v16bf v; uint4 q[2]; };

#if defined(__has_builtin)
#if __has_builtin(__builtin_amdgcn_global_load_async_to_lds_b128)
#define HAVE_ASYNC_B128 1
#endif
#if __has_builtin(__builtin_amdgcn_s_wait_asynccnt)
#define HAVE_WAIT_ASYNC 1
#endif
#endif

// global -> LDS async copy, 16B per lane, tracked by ASYNCcnt
static __device__ inline void async_b128(const void* g, void* l) {
#ifdef HAVE_ASYNC_B128
    __builtin_amdgcn_global_load_async_to_lds_b128(
        (AS1 v4i*)(AS1 void*)(void*)g,
        (AS3 v4i*)(AS3 void*)l, 0, 0);
#elif defined(__AMDGCN__)
    unsigned laddr = (unsigned)(unsigned long long)(AS3 void*)l;  // 32-bit LDS byte address
    unsigned long long gaddr = (unsigned long long)(uintptr_t)g;
    asm volatile("global_load_async_to_lds_b128 %0, %1, off"
                 :: "v"(laddr), "v"(gaddr) : "memory");
#else
    (void)g; (void)l;   // host pass: never executed
#endif
}

static __device__ inline void wait_async0() {
#ifdef HAVE_WAIT_ASYNC
    __builtin_amdgcn_s_wait_asynccnt(0);
#elif defined(__AMDGCN__)
    asm volatile("s_wait_asynccnt 0x0" ::: "memory");
#endif
}

// ---------------------------------------------------------------------------
// prep_x: Xs[m,k] = bf16( x[m,k] * scaling0[k] )
// ---------------------------------------------------------------------------
__global__ __launch_bounds__(256)
void prep_x(const float* __restrict__ x, const float* __restrict__ s0,
            __bf16* __restrict__ xs)
{
    size_t i = ((size_t)blockIdx.x * 256 + threadIdx.x) * 4;
    const float4 xv = *(const float4*)(x + i);
    const int k = (int)(i & (K_DIM - 1));
    const float4 sv = *(const float4*)(s0 + k);
    union { __bf16 h[4]; uint2 u; } o;
    o.h[0] = (__bf16)(xv.x * sv.x);
    o.h[1] = (__bf16)(xv.y * sv.y);
    o.h[2] = (__bf16)(xv.z * sv.z);
    o.h[3] = (__bf16)(xv.w * sv.w);
    *(uint2*)(xs + i) = o.u;
}

// ---------------------------------------------------------------------------
// prep_w: W[n,k] = bit ? +1 : -1  (bf16 patterns 0x3F80 / 0xBF80)
// ---------------------------------------------------------------------------
__global__ __launch_bounds__(256)
void prep_w(const int* __restrict__ bits, __bf16* __restrict__ w)
{
    size_t i = ((size_t)blockIdx.x * 256 + threadIdx.x) * 4;
    const int4 b = *(const int4*)(bits + i);
    union { u16 h[4]; uint2 u; } o;
    o.h[0] = b.x ? 0x3F80u : 0xBF80u;
    o.h[1] = b.y ? 0x3F80u : 0xBF80u;
    o.h[2] = b.z ? 0x3F80u : 0xBF80u;
    o.h[3] = b.w ? 0x3F80u : 0xBF80u;
    *(uint2*)(w + i) = o.u;
}

// ---------------------------------------------------------------------------
// gemm_bin: C[m,n] = (sum_k A[m,k]*Bw[n,k]) * scale[n] (+ bias[n])
// Block 128x128, BK=64, 8 waves (4Mx2N), wave tile 32x64, async global->LDS.
// ---------------------------------------------------------------------------
template<bool FINAL>
__global__ __launch_bounds__(NTHREADS)
void gemm_bin(const __bf16* __restrict__ A,
              const __bf16* __restrict__ Bw,
              const float*  __restrict__ scale,
              const float*  __restrict__ bias,
              __bf16* __restrict__ outB,
              float*  __restrict__ outF,
              int K, int N)
{
    __shared__ __align__(16) u16 ldsA[2][BM * LDA];
    __shared__ __align__(16) u16 ldsB[2][BN * LDA];

    const int tid    = threadIdx.x;
    const int lane   = tid & 31;
    const int wid    = tid >> 5;
    const int half   = lane >> 4;
    const int l16    = lane & 15;
    const int wave_m = (wid >> 1) * 32;
    const int wave_n = (wid & 1)  * 64;
    const int bm     = blockIdx.y * BM;
    const int bn     = blockIdx.x * BN;

    // staging map: 1024 16B chunks/tile; thread owns col-chunk (tid&7),
    // rows (tid>>3) + 32*i, i = 0..3
    const int crow = tid >> 3;
    const int ccol = tid & 7;

    const __bf16* gA = A  + (size_t)(bm + crow) * K + ccol * 8;
    const __bf16* gB = Bw + (size_t)(bn + crow) * K + ccol * 8;
    const size_t rowStep = (size_t)32 * K;     // 32 rows (bf16 elems)

    v8f acc[2][4] = {};
    const int NK = K / BK;

    // ---- prologue: async-fill buffer 0 ----
    {
#pragma unroll
        for (int i = 0; i < 4; ++i) {
            async_b128(gA + rowStep * i, &ldsA[0][(crow + 32 * i) * LDA + ccol * 8]);
            async_b128(gB + rowStep * i, &ldsB[0][(crow + 32 * i) * LDA + ccol * 8]);
        }
        gA += BK; gB += BK;
    }
    wait_async0();
    __syncthreads();

    for (int kt = 0; kt < NK; ++kt) {
        const int cur = kt & 1;

        // ---- async prefetch next stage into other buffer ----
        if (kt + 1 < NK) {
            const int nxt = cur ^ 1;
#pragma unroll
            for (int i = 0; i < 4; ++i) {
                async_b128(gA + rowStep * i, &ldsA[nxt][(crow + 32 * i) * LDA + ccol * 8]);
                async_b128(gB + rowStep * i, &ldsB[nxt][(crow + 32 * i) * LDA + ccol * 8]);
            }
            gA += BK; gB += BK;
        }

        // ---- compute current stage: 2 WMMA k-steps of 32 ----
#pragma unroll
        for (int ks = 0; ks < 2; ++ks) {
            const int kk = ks * 32;

            FragAB aF[2];
#pragma unroll
            for (int i = 0; i < 2; ++i) {
                const int r = wave_m + i * 16 + l16;
                const u16* p = &ldsA[cur][r * LDA + kk + half * 8];
                aF[i].q[0] = *(const uint4*)(p);       // K = kk + half*8 .. +7
                aF[i].q[1] = *(const uint4*)(p + 16);  // K = kk+16+half*8 .. +7
            }

            FragAB bF[4];
#pragma unroll
            for (int j = 0; j < 4; ++j) {
                const int n = wave_n + j * 16 + l16;
                const u16* p = &ldsB[cur][n * LDA + kk + half * 8];
                bF[j].q[0] = *(const uint4*)(p);       // K = kk + half*8 .. +7
                bF[j].q[1] = *(const uint4*)(p + 16);  // K = kk+16+half*8 .. +7
            }

#pragma unroll
            for (int i = 0; i < 2; ++i)
#pragma unroll
                for (int j = 0; j < 4; ++j)
                    acc[i][j] = __builtin_amdgcn_wmma_f32_16x16x32_bf16(
                        false, aF[i].v, false, bF[j].v,
                        (short)0, acc[i][j], false, false);
        }

        if (kt + 1 < NK) wait_async0();
        __syncthreads();
    }

    // ---- epilogue: C/D layout: VGPR r -> M = r + 8*half, lane&15 -> N ----
#pragma unroll
    for (int i = 0; i < 2; ++i) {
#pragma unroll
        for (int j = 0; j < 4; ++j) {
            const int gn = bn + wave_n + j * 16 + l16;
            const float s = scale[gn];
            const float b = FINAL ? bias[gn] : 0.0f;
            const int gm0 = bm + wave_m + i * 16 + half * 8;
#pragma unroll
            for (int r = 0; r < 8; ++r) {
                const float v = acc[i][j][r] * s + b;
                const size_t off = (size_t)(gm0 + r) * N + gn;
                if (FINAL) outF[off] = v;
                else       outB[off] = (__bf16)v;
            }
        }
    }
}

// ---------------------------------------------------------------------------
// launch
// ---------------------------------------------------------------------------
extern "C" void kernel_launch(void* const* d_in, const int* in_sizes, int n_in,
                              void* d_out, int out_size, void* d_ws, size_t ws_size,
                              hipStream_t stream)
{
    const float* x    = (const float*)d_in[0];
    const float* s0   = (const float*)d_in[1];
    const float* s2   = (const float*)d_in[2];
    const float* s4   = (const float*)d_in[3];
    const float* bias = (const float*)d_in[4];
    const int*   w1b  = (const int*)d_in[5];
    const int*   w3b  = (const int*)d_in[6];
    float* out = (float*)d_out;

    char* ws = (char*)d_ws;
    const size_t XS_BYTES = (size_t)M_TOT * K_DIM * 2;   // 64 MiB
    const size_t W_BYTES  = (size_t)N_DIM * K_DIM * 2;   // 32 MiB each
    __bf16* Xs  = (__bf16*)(ws);
    __bf16* W1  = (__bf16*)(ws + XS_BYTES);
    __bf16* W3  = (__bf16*)(ws + XS_BYTES + W_BYTES);
    __bf16* Hb  = (__bf16*)(ws + XS_BYTES + 2 * W_BYTES);

    prep_x<<<(M_TOT * (size_t)K_DIM / 4) / 256, 256, 0, stream>>>(x, s0, Xs);
    prep_w<<<((size_t)N_DIM * K_DIM / 4) / 256, 256, 0, stream>>>(w1b, W1);
    prep_w<<<((size_t)N_DIM * K_DIM / 4) / 256, 256, 0, stream>>>(w3b, W3);

    dim3 grid(N_DIM / BN, M_TOT / BM);
    gemm_bin<false><<<grid, NTHREADS, 0, stream>>>(Xs, W1, s2, nullptr, Hb, nullptr, K_DIM, N_DIM);
    gemm_bin<true><<<grid, NTHREADS, 0, stream>>>(Hb, W3, s4, bias, nullptr, out, K_DIM, N_DIM);
}
